// OscillatorBank_12249246728370
// MI455X (gfx1250) — matrix operand order repaired and checked by reading, not compiled, not run on hardware
//
#include <hip/hip_runtime.h>

// ============================ problem constants ============================
#define BB 8      // batch
#define CC 128    // in channels (K)
#define TT 128    // frames (N of gemm)
#define OO 128    // oscillators (M of gemm)
#define NA 32768  // audio samples (upsample x256)

typedef __attribute__((ext_vector_type(16))) _Float16 v16h;
typedef __attribute__((ext_vector_type(8)))  _Float16 v8h;
typedef __attribute__((ext_vector_type(8)))  float    v8f;

#define LDS_PITCH (CC + 8)   // +16B pad: fragment reads hit distinct LDS banks

// ============================================================================
// Kernel 1: 1x1 conv (= GEMM  out[o,t] = sum_c W[o,c] * X[c,t]) + bias +
// sigmoid via v_wmma_f32_16x16x32_f16. grid = (batch, amp/freq), block = 256
// (8 waves). X[b] is staged ONCE per block into LDS, transposed + converted
// to f16, so each B fragment is two ds_load_b128 instead of 16 strided global
// loads, and all 8 waves share the staged copy. Wave w owns output tile-row w
// and sweeps the 8 tile-columns reusing its A fragment.
// ============================================================================
__global__ __launch_bounds__(256) void osc_gemm_sigmoid(
    const float* __restrict__ x,
    const float* __restrict__ amp_w,  const float* __restrict__ amp_b,
    const float* __restrict__ freq_w, const float* __restrict__ freq_b,
    float* __restrict__ amp_sig, float* __restrict__ freq_sig)
{
  __shared__ _Float16 xsT[TT][LDS_PITCH];  // X transposed: xsT[t][c], ~34 KB

  const int  b   = blockIdx.x;
  const bool isf = (blockIdx.y != 0);
  const float* W   = isf ? freq_w : amp_w;
  const float* Bi  = isf ? freq_b : amp_b;
  float*       Out = isf ? freq_sig : amp_sig;

  const int lane = threadIdx.x & 31;
  const int wv   = threadIdx.x >> 5;   // wave id == M tile
  const int half = lane >> 4;          // 0: lanes0-15, 1: lanes16-31
  const int lm   = lane & 15;

  const float* xb = x + b * CC * TT;

  // ---- stage X[b]: coalesced global_load_b128, transpose+cvt into LDS ----
  for (int i = threadIdx.x; i < (CC * TT) / 4; i += 256) {
    const float4 v = ((const float4*)xb)[i];
    const int flat = i * 4;           // = c*TT + t
    const int c = flat >> 7;
    const int t = flat & (TT - 1);    // multiple of 4, stays in row c
    xsT[t + 0][c] = (_Float16)v.x;
    xsT[t + 1][c] = (_Float16)v.y;
    xsT[t + 2][c] = (_Float16)v.z;
    xsT[t + 3][c] = (_Float16)v.w;
  }
  __syncthreads();

  v8f acc[8] = {};

  #pragma unroll
  for (int kk = 0; kk < CC; kk += 32) {
    // A fragment (16x32 f16): lane holds row m = wv*16+lm.
    // ISA layout: element e -> k = (e/8)*16 + half*8 + (e%8)   (§7.12.2)
    v16h afrag;
    {
      const float4* w4 =
          (const float4*)(W + (wv * 16 + lm) * CC + kk + half * 8);
      const float4 a0 = w4[0], a1 = w4[1];   // k: +0..7
      const float4 a2 = w4[4], a3 = w4[5];   // k: +16..23
      afrag[0]  = (_Float16)a0.x; afrag[1]  = (_Float16)a0.y;
      afrag[2]  = (_Float16)a0.z; afrag[3]  = (_Float16)a0.w;
      afrag[4]  = (_Float16)a1.x; afrag[5]  = (_Float16)a1.y;
      afrag[6]  = (_Float16)a1.z; afrag[7]  = (_Float16)a1.w;
      afrag[8]  = (_Float16)a2.x; afrag[9]  = (_Float16)a2.y;
      afrag[10] = (_Float16)a2.z; afrag[11] = (_Float16)a2.w;
      afrag[12] = (_Float16)a3.x; afrag[13] = (_Float16)a3.y;
      afrag[14] = (_Float16)a3.z; afrag[15] = (_Float16)a3.w;
    }
    #pragma unroll
    for (int tn = 0; tn < 8; ++tn) {
      // B fragment (32x16 f16): lane holds column n = tn*16+lm.
      // ISA layout: element e -> k = (e/8)*16 + half*8 + (e%8) (matches A).
      // Contiguous in xsT[n][k] -> two 16B ds loads, bank-conflict-free.
      const _Float16* bp = &xsT[tn * 16 + lm][kk + half * 8];
      const v8h lo = *(const v8h*)bp;
      const v8h hi = *(const v8h*)(bp + 16);
      const v16h bfrag = __builtin_shufflevector(
          lo, hi, 0, 1, 2, 3, 4, 5, 6, 7, 8, 9, 10, 11, 12, 13, 14, 15);

      acc[tn] = __builtin_amdgcn_wmma_f32_16x16x32_f16(
          false, afrag, false, bfrag, (short)0, acc[tn], false, false);
    }
  }

  // C/D layout: VGPR r -> M = half*8 + r, N = lm (§7.12.2). Fused bias+sigmoid.
  #pragma unroll
  for (int tn = 0; tn < 8; ++tn) {
    #pragma unroll
    for (int r = 0; r < 8; ++r) {
      const int o = wv * 16 + half * 8 + r;
      const int t = tn * 16 + lm;
      const float v = acc[tn][r] + Bi[o];
      Out[(b * OO + o) * TT + t] = 1.0f / (1.0f + __expf(-v));
    }
  }
}

// ============================================================================
// Kernel 2: phase prefix in "turns" (phase / 2pi), mod-1 reduced.
// Upsample x256, align_corners=False => sample n=256t+128+m has w=(m+0.5)/256.
// Segment total (turns) = 64*(f_t + f_{t+1}); head (128 clipped samples)= 64*f0.
// pref[t] = turns accumulated *before* segment t (t=0..126); pref[127] feeds
// the clipped tail. 1024 independent 128-length scans (one thread each).
// ============================================================================
__global__ __launch_bounds__(256) void osc_phase_prefix(
    const float* __restrict__ freq_sig, float* __restrict__ pref)
{
  const int id = blockIdx.x * blockDim.x + threadIdx.x;  // b*128 + o
  if (id >= BB * OO) return;
  const float* f = freq_sig + id * TT;
  float* p = pref + id * TT;

  const float f0 = f[0];
  float acc = 64.0f * f0;            // head total
  acc -= rintf(acc);                 // sin(2pi x) is 1-periodic: keep |acc| small
  p[0] = acc;
  float prev = f0;
  for (int t = 1; t < TT; ++t) {
    const float ft = f[t];
    acc += 64.0f * (prev + ft);
    acc -= rintf(acc);
    p[t] = acc;
    prev = ft;
  }
}

// ============================================================================
// Kernel 3: additive synthesis. grid = (segment s in 0..128, batch).
//  s=0   : head, samples 0..127        (freq/amp clipped to frame 0)
//  s=1..127: segment t=s-1, samples 256t+128 .. +255
//  s=128 : tail, samples 32640..32767  (clipped to frame 127)
// Per-sample phase (turns) = pref + 0.5*(m+1)*f0 + (f1-f0)*(m+1)^2/1024.
// v_sin_f32 natively computes sin(2pi x) -> feed fract-reduced turns directly.
// ============================================================================
__global__ __launch_bounds__(256) void osc_synth(
    const float* __restrict__ freq_sig, const float* __restrict__ amp_sig,
    const float* __restrict__ pref, float* __restrict__ out)
{
  __shared__ float sf0[OO], sf1[OO], sa0[OO], sa1[OO], sp[OO];

  const int b = blockIdx.y;
  const int s = blockIdx.x;
  const int tid = threadIdx.x;

  if (tid < OO) {
    const int o = tid;
    const float* fb = freq_sig + (b * OO + o) * TT;
    const float* ab = amp_sig  + (b * OO + o) * TT;
    const float* pb = pref     + (b * OO + o) * TT;
    if (s == 0)        { sf0[o]=fb[0];     sf1[o]=fb[0];     sa0[o]=ab[0];     sa1[o]=ab[0];     sp[o]=0.0f;    }
    else if (s == 128) { sf0[o]=fb[TT-1];  sf1[o]=fb[TT-1];  sa0[o]=ab[TT-1];  sa1[o]=ab[TT-1];  sp[o]=pb[TT-1];}
    else { const int t = s - 1;
           sf0[o]=fb[t]; sf1[o]=fb[t+1];   sa0[o]=ab[t];     sa1[o]=ab[t+1];   sp[o]=pb[t];     }
  }
  __syncthreads();

  int n;
  bool active = true;
  if (s == 0)        { n = tid;                       active = tid < 128; }
  else if (s == 128) { n = (NA - 128) + tid;          active = tid < 128; }
  else               { n = (s - 1) * 256 + 128 + tid; }
  if (!active) return;

  const float mm   = (float)tid + 1.0f;
  const float wamp = ((float)tid + 0.5f) * (1.0f / 256.0f);
  const float q    = mm * mm * (1.0f / 1024.0f);
  const float h    = 0.5f * mm;

  float acc = 0.0f;
  #pragma unroll 4
  for (int o = 0; o < OO; ++o) {
    const float f0 = sf0[o];
    const float df = sf1[o] - f0;
    float turns = sp[o] + h * f0 + df * q;
    turns -= rintf(turns);                          // v_rndne: [-0.5,0.5]
    const float sv = __builtin_amdgcn_sinf(turns);  // v_sin_f32: sin(2*pi*x)
    const float a0 = sa0[o];
    acc = __builtin_fmaf(a0 + wamp * (sa1[o] - a0), sv, acc);
  }
  out[b * NA + n] = acc * (1.0f / (float)OO);
}

// ============================ host-side launch =============================
extern "C" void kernel_launch(void* const* d_in, const int* in_sizes, int n_in,
                              void* d_out, int out_size, void* d_ws, size_t ws_size,
                              hipStream_t stream) {
  const float* x      = (const float*)d_in[0];
  const float* amp_w  = (const float*)d_in[1];
  const float* amp_b  = (const float*)d_in[2];
  const float* freq_w = (const float*)d_in[3];
  const float* freq_b = (const float*)d_in[4];
  float* out = (float*)d_out;

  // workspace: 3 x [B,O,T] f32 = 3 x 512 KB
  float* ws       = (float*)d_ws;
  float* amp_sig  = ws;
  float* freq_sig = ws + (BB * OO * TT);
  float* pref     = ws + 2 * (BB * OO * TT);

  dim3 g1(BB, 2);
  osc_gemm_sigmoid<<<g1, 256, 0, stream>>>(x, amp_w, amp_b, freq_w, freq_b,
                                           amp_sig, freq_sig);

  osc_phase_prefix<<<(BB * OO + 255) / 256, 256, 0, stream>>>(freq_sig, pref);

  dim3 g3(129, BB);
  osc_synth<<<g3, 256, 0, stream>>>(freq_sig, amp_sig, pref, out);
}
